// Cormorant_34333968564813
// MI455X (gfx1250) — matrix operand order repaired and checked by reading, not compiled, not run on hardware
//
#include <hip/hip_runtime.h>
#include <hip/hip_bf16.h>

// ---------------------------------------------------------------------------
// Cormorant forward for MI455X (gfx1250, wave32, WMMA).
// All C-dimension GEMMs mapped to v_wmma_f32_16x16x32_f16 (f16 in, f32 acc).
// Weights pre-converted to f16 (K-padded); every WMMA fragment is built from
// contiguous 16B vector loads. sph tiles staged to LDS with CDNA5
// global_load_async_to_lds_b128 (ASYNCcnt) in the aggregation kernel.
// ---------------------------------------------------------------------------

typedef __attribute__((ext_vector_type(16))) _Float16 v16h;
typedef __attribute__((ext_vector_type(8)))  _Float16 v8h;
typedef __attribute__((ext_vector_type(8)))  float    v8f;

static constexpr int B_    = 16;
static constexpr int N_    = 64;
static constexpr int C_    = 32;
static constexpr int LVLS_ = 4;
static constexpr int MAXL_ = 3;
static constexpr int NB_   = 60;     // NBASIS
static constexpr int KPAD_ = 64;     // basis K padded to 64 (two k-steps of 32)
static constexpr long PAIRS_ = (long)B_ * N_ * N_;   // 65536
static constexpr int SLOTS_ = 16;    // 1+3+5+7 spherical slots

#define WMMA_F16(A, Bm, Cacc) \
  __builtin_amdgcn_wmma_f32_16x16x32_f16(false, (A), false, (Bm), (short)0, (Cacc), false, false)

__device__ __forceinline__ v16h cat16(v8h lo, v8h hi) {
  return __builtin_shufflevector(lo, hi, 0,1,2,3,4,5,6,7,8,9,10,11,12,13,14,15);
}
// 16 contiguous f16 (32B) -> fragment register pair
__device__ __forceinline__ v16h ld16(const _Float16* __restrict__ p) {
  return cat16(*(const v8h*)p, *(const v8h*)(p + 8));
}
// A-fragment row load: ISA A layout per lane-group g is two contiguous 8-half
// runs at k = 8g..8g+7 and k = 16+8g..16+8g+7.
__device__ __forceinline__ v16h ldA(const _Float16* __restrict__ row, int g) {
  return cat16(*(const v8h*)(row + 8 * g), *(const v8h*)(row + 16 + 8 * g));
}

// CDNA5 async global->LDS copy (16B). The LDS offset is derived from the real
// shared-memory pointer (low 32 bits of the generic address = LDS offset), so
// the destination escapes into the asm and loads from it stay well-defined.
__device__ __forceinline__ void asyncLdsB128(void* ldsDst, const void* gsrc) {
  unsigned loff = (unsigned)(unsigned long long)(uintptr_t)ldsDst;
  unsigned long long ga = (unsigned long long)(uintptr_t)gsrc;
  asm volatile("global_load_async_to_lds_b128 %0, %1, off"
               :: "v"(loff), "v"(ga) : "memory");
}
__device__ __forceinline__ void waitAsync() {
  asm volatile("s_wait_asynccnt 0" ::: "memory");
}

// ---------------------------------------------------------------------------
// Prep 1: a0 = (scalars_in @ W_in + b_in) * amask ; init reps[0] slabs
// (f32 slab for head, f16 shadow slab for WMMA A-operands). thread=(b,n,c)
// ---------------------------------------------------------------------------
__global__ __launch_bounds__(256)
void kPrepAtoms(const float* __restrict__ one_hot, const int* __restrict__ charges,
                const unsigned char* __restrict__ amask,
                const float* __restrict__ W_in, const float* __restrict__ b_in,
                float* __restrict__ reps0, _Float16* __restrict__ repsH0)
{
  int t = blockIdx.x * blockDim.x + threadIdx.x;
  if (t >= B_ * N_ * C_) return;
  int c  = t & (C_ - 1);
  int bn = t >> 5;
  float ch = (float)charges[bn] / 9.0f;          // CHARGE_SCALE
  float pw1 = ch, pw2 = ch * ch;
  float acc = b_in[c];
  #pragma unroll
  for (int s = 0; s < 5; ++s) {
    float oh = one_hot[bn * 5 + s];
    const float* w = W_in + (s * 3) * C_ + c;
    acc += oh * (w[0] + pw1 * w[C_] + pw2 * w[2 * C_]);
  }
  acc *= (float)amask[bn];
  float*    r0 = reps0  + (long)bn * SLOTS_ * C_;
  _Float16* rH = repsH0 + (long)bn * SLOTS_ * C_;
  r0[c] = acc;
  rH[c] = (_Float16)acc;
  #pragma unroll
  for (int s = 1; s < SLOTS_; ++s) { r0[s * C_ + c] = 0.0f; rH[s * C_ + c] = (_Float16)0.0f; }
}

// ---------------------------------------------------------------------------
// Prep W: one-shot f32 -> f16 weight conversion. W_rad K-padded 60 -> 64
// with zero rows so the k-step-1 fragment needs no predication.
// ---------------------------------------------------------------------------
__global__ __launch_bounds__(256)
void kPrepW(const float* __restrict__ W_rad, const float* __restrict__ W_edge,
            const float* __restrict__ W_agg, const float* __restrict__ W_self,
            _Float16* __restrict__ wRadH, _Float16* __restrict__ wEdgeH,
            _Float16* __restrict__ wAggH, _Float16* __restrict__ wSelfH)
{
  int t = blockIdx.x * blockDim.x + threadIdx.x;
  if (t < 16 * KPAD_ * C_) {                     // 16 mats x 64 x 32
    int k = t & 63, c = (t >> 6) & 31, mat = t >> 11;
    wRadH[t] = (k < NB_) ? (_Float16)W_rad[((long)mat * NB_ + k) * C_ + c]
                         : (_Float16)0.0f;
  }
  if (t < 16 * C_ * C_) {                        // 16 mats x 32 x 32
    int src = ((t >> 10) << 10) | ((t & 31) << 5) | ((t >> 5) & 31);
    wEdgeH[t] = (_Float16)W_edge[src];
    wAggH[t]  = (_Float16)W_agg[src];
    wSelfH[t] = (_Float16)W_self[src];
  }
}

// ---------------------------------------------------------------------------
// Prep 2: per-pair geometry: sph (16 comps), cutoff, f16 radial basis (padded
// to 64), and zero-init the 4 edge_net buffers. one thread per (b,i,j)
// ---------------------------------------------------------------------------
__global__ __launch_bounds__(256)
void kPrepPairs(const float* __restrict__ pos, const unsigned char* __restrict__ emask,
                _Float16* __restrict__ basisH, float* __restrict__ sph,
                float* __restrict__ cutoff, float* __restrict__ edge)
{
  long p = (long)blockIdx.x * blockDim.x + threadIdx.x;
  if (p >= PAIRS_) return;
  int j  = (int)(p & (N_ - 1));
  long bi = p >> 6;
  int i  = (int)(bi & (N_ - 1));
  int b  = (int)(bi >> 6);
  const float* pi = pos + ((long)b * N_ + i) * 3;
  const float* pj = pos + ((long)b * N_ + j) * 3;
  float rx = pi[0] - pj[0], ry = pi[1] - pj[1], rz = pi[2] - pj[2];
  float nrm = sqrtf(rx * rx + ry * ry + rz * rz + 1e-12f);
  float inv = 1.0f / nrm;
  float x = rx * inv, y = ry * inv, z = rz * inv;
  float* sp = sph + p * SLOTS_;
  sp[0]  = 1.0f;
  sp[1]  = x;                         sp[2]  = y;            sp[3] = z;
  sp[4]  = x * y;                     sp[5]  = y * z;
  sp[6]  = 3.0f * z * z - 1.0f;       sp[7]  = x * z;        sp[8] = x * x - y * y;
  sp[9]  = y * (3.0f * x * x - y * y);
  sp[10] = x * y * z;
  sp[11] = y * (5.0f * z * z - 1.0f);
  sp[12] = z * (5.0f * z * z - 3.0f);
  sp[13] = x * (5.0f * z * z - 1.0f);
  sp[14] = z * (x * x - y * y);
  sp[15] = x * (x * x - 3.0f * y * y);
  float valid = (emask[p] ? 1.0f : 0.0f) * (nrm > 1e-4f ? 1.0f : 0.0f);
  float sg = 1.0f / (1.0f + __expf(-(3.0f - nrm) / 0.5f));   // sigmoid((SOFT-r)/W)
  cutoff[p] = valid * (nrm < 5.0f ? 1.0f : 0.0f) * sg;
  _Float16* bp = basisH + p * KPAD_;
  #pragma unroll
  for (int k = 0; k < KPAD_; ++k) {
    if (k < NB_) {
      float d = nrm - 5.0f * (float)k / 59.0f;               // linspace(0,RMAX,60)
      bp[k] = (_Float16)__expf(-36.0f * d * d);              // gamma = (60/5)^2/4
    } else bp[k] = (_Float16)0.0f;
  }
  float* eb = edge + p * C_;
  #pragma unroll
  for (int l = 0; l < 4; ++l)
    for (int c = 0; c < C_; ++c) eb[(long)l * PAIRS_ * C_ + c] = 0.0f;
}

// ---------------------------------------------------------------------------
// Edge kernel: one wave per (b,i, jtile of 16 rows). Per l:
//   rad  = basis(16x64) @ W_rad  (4 WMMA)
//   e    = s_ij(16x32)  @ W_edge (2 WMMA)
//   edge[l] = cutoff * rad * (e + edge_prev[l])     (in-place f32)
// All fragments are contiguous 16B f16 vector loads. Each lane loads its 8
// needed cutoff values directly as one 32B-aligned v8f (no LDS, no barrier).
// ---------------------------------------------------------------------------
__global__ __launch_bounds__(32)
void kEdge(const _Float16* __restrict__ repsHIn,   // [B*N,16,C] f16, slot0 = scalar rep
           const _Float16* __restrict__ basisH, const float* __restrict__ cutoff,
           float* __restrict__ edge,
           const _Float16* __restrict__ wRadH,     // [16][32][64] f16 ([n][k]), K-padded
           const _Float16* __restrict__ wEdgeH,    // [16][32][32] f16
           int lvl)
{
  int wid = blockIdx.x;                // (b*N + i)*4 + jt
  int jt  = wid & 3;
  long bi = wid >> 2;                  // b*N + i
  int  b  = (int)(bi >> 6);
  int  j0 = jt * 16;
  int lane = threadIdx.x;
  int lm = lane & 15;                  // A-fragment row / D-fragment column
  int g  = lane >> 4;
  long eRowBase = bi * N_ + j0;        // pair index of tile row 0

  // cutoff: rows rm = 8g..8g+7 for this lane -> one contiguous 32B run
  v8f cutV = *(const v8f*)(cutoff + eRowBase + 8 * g);

  // A fragment: s_ij rows (row m -> j = j0+m), packed f16 product
  const _Float16* ai = repsHIn + bi * SLOTS_ * C_;                        // slot 0
  const _Float16* aj = repsHIn + ((long)b * N_ + (j0 + lm)) * SLOTS_ * C_;
  v8h ai0 = *(const v8h*)(ai + 8 * g), ai1 = *(const v8h*)(ai + 16 + 8 * g);
  v8h aj0 = *(const v8h*)(aj + 8 * g), aj1 = *(const v8h*)(aj + 16 + 8 * g);
  v16h aS = cat16(ai0 * aj0, ai1 * aj1);          // v_pk_mul_f16

  // A fragments: radial basis 16x64 -> two 16x32 k-steps (contiguous 16B runs)
  const _Float16* bp = basisH + (eRowBase + lm) * KPAD_;
  v16h aB0 = ldA(bp, g);
  v16h aB1 = ldA(bp + 32, g);

  __builtin_prefetch(wRadH  + (long)(lvl * 4) * KPAD_ * C_, 0, 1);  // global_prefetch_b8
  __builtin_prefetch(wEdgeH + (long)(lvl * 4) * C_ * C_,   0, 1);

  for (int l = 0; l <= MAXL_; ++l) {
    const _Float16* wr = wRadH  + (long)(lvl * 4 + l) * KPAD_ * C_;
    const _Float16* we = wEdgeH + (long)(lvl * 4 + l) * C_ * C_;
    // B fragments: lane = N, elements = K in table order (weights stored [n][k])
    v16h bR00 = ldA(wr + (long)lm * KPAD_, g);
    v16h bR01 = ldA(wr + (long)(16 + lm) * KPAD_, g);
    v16h bR10 = ldA(wr + (long)lm * KPAD_ + 32, g);        // rows 60..63 pre-zeroed
    v16h bR11 = ldA(wr + (long)(16 + lm) * KPAD_ + 32, g);
    v16h bE0  = ldA(we + (long)lm * C_, g);
    v16h bE1  = ldA(we + (long)(16 + lm) * C_, g);

    v8f rad0 = {}, rad1 = {}, e0 = {}, e1 = {};
    rad0 = WMMA_F16(aB0, bR00, rad0);
    rad0 = WMMA_F16(aB1, bR10, rad0);
    rad1 = WMMA_F16(aB0, bR01, rad1);
    rad1 = WMMA_F16(aB1, bR11, rad1);
    e0   = WMMA_F16(aS,  bE0,  e0);
    e1   = WMMA_F16(aS,  bE1,  e1);

    float* eb = edge + (long)l * PAIRS_ * C_;
    #pragma unroll
    for (int v = 0; v < 8; ++v) {                 // D layout: row = v + 8*g, col = lm
      int rm = v + 8 * g;
      long idx = (eRowBase + rm) * C_ + lm;
      float cu = cutV[v];
      float o0 = eb[idx], o1 = eb[idx + 16];
      eb[idx]      = cu * rad0[v] * (e0[v] + o0);
      eb[idx + 16] = cu * rad1[v] * (e1[v] + o1);
    }
  }
}

// ---------------------------------------------------------------------------
// Aggregation: msg[b,i,slot,c] = sum_j edge[l(slot)][b,i,j,c]*sph[slot]*amask[j]
// sph tile (64 x 16 f32 = 4KB) staged to LDS via CDNA5 async-to-LDS; each
// staged value is reused by all 32 channels. msg stored directly in f16.
// ---------------------------------------------------------------------------
__global__ __launch_bounds__(128)
void kMsg(const float* __restrict__ edge, const float* __restrict__ sph,
          const unsigned char* __restrict__ amask, _Float16* __restrict__ msgH)
{
  __shared__ float sphS[N_ * SLOTS_];   // 4 KB
  long bi = blockIdx.x;                 // b*N + i
  int  b  = (int)(bi >> 6);
  int  t  = threadIdx.x;
  long pbase = bi * N_;

  // async stage: 256 chunks of 16B, 2 per thread (GLOBAL_LOAD_ASYNC_TO_LDS_B128)
  const float* sphSrc = sph + pbase * SLOTS_;
  #pragma unroll
  for (int q = 0; q < 2; ++q) {
    int chunk = t + q * 128;
    asyncLdsB128(&sphS[chunk * 4], sphSrc + chunk * 4);
  }
  waitAsync();
  __syncthreads();

  int c  = t & (C_ - 1);
  int sg = t >> 5;                      // 4 slot-groups of 4
  #pragma unroll
  for (int ss = 0; ss < 4; ++ss) {
    int s = sg * 4 + ss;
    int l = (s == 0) ? 0 : (s < 4) ? 1 : (s < 9) ? 2 : 3;
    const float* eb = edge + (long)l * PAIRS_ * C_;
    float acc = 0.0f;
    for (int j = 0; j < N_; ++j) {
      long p = pbase + j;
      acc += eb[p * C_ + c] * sphS[j * SLOTS_ + s] * (float)amask[b * N_ + j];
    }
    msgH[(bi * SLOTS_ + s) * C_ + c] = (_Float16)acc;
  }
}

// ---------------------------------------------------------------------------
// Rep kernel: rep = msg @ W_agg[lvl,l] + repsIn @ W_self[lvl,l], * amask[i].
// Rows = (b,i,m) flattened per l, tiled by 16; one wave per tile, 4 WMMAs.
// A operands are pure f16 vector loads (msgH + f16 reps shadow).
// ---------------------------------------------------------------------------
__global__ __launch_bounds__(32)
void kRep(const _Float16* __restrict__ msgH, const _Float16* __restrict__ repsHIn,
          float* __restrict__ repsOut, _Float16* __restrict__ repsHOut,
          const _Float16* __restrict__ wAggH, const _Float16* __restrict__ wSelfH,
          const unsigned char* __restrict__ amask, int lvl)
{
  int tid = blockIdx.x;                 // 1024 tiles: 64*(1+3+5+7)
  int l, tl;
  if      (tid < 64)  { l = 0; tl = tid; }
  else if (tid < 256) { l = 1; tl = tid - 64; }
  else if (tid < 576) { l = 2; tl = tid - 256; }
  else                { l = 3; tl = tid - 576; }
  int nm = 2 * l + 1;
  int sb = l * l;                       // slot base 0,1,4,9
  int lane = threadIdx.x;
  int lm = lane & 15;
  int g  = lane >> 4;

  int r  = tl * 16 + lm;                // A-fragment row
  int bi = r / nm, m = r % nm;
  v16h aM = ldA(msgH    + ((long)bi * SLOTS_ + sb + m) * C_, g);
  v16h aR = ldA(repsHIn + ((long)bi * SLOTS_ + sb + m) * C_, g);

  const _Float16* wa = wAggH  + (long)(lvl * 4 + l) * C_ * C_;
  const _Float16* ws = wSelfH + (long)(lvl * 4 + l) * C_ * C_;
  v16h bA0 = ldA(wa + (long)lm * C_, g);
  v16h bA1 = ldA(wa + (long)(16 + lm) * C_, g);
  v16h bS0 = ldA(ws + (long)lm * C_, g);
  v16h bS1 = ldA(ws + (long)(16 + lm) * C_, g);

  v8f d0 = {}, d1 = {};
  d0 = WMMA_F16(aM, bA0, d0);
  d0 = WMMA_F16(aR, bS0, d0);
  d1 = WMMA_F16(aM, bA1, d1);
  d1 = WMMA_F16(aR, bS1, d1);

  #pragma unroll
  for (int v = 0; v < 8; ++v) {
    int rr = tl * 16 + v + 8 * g;       // global row of this D element
    int bi2 = rr / nm, m2 = rr % nm;
    float am = (float)amask[bi2];
    long o = ((long)bi2 * SLOTS_ + sb + m2) * C_;
    float y0 = am * d0[v], y1 = am * d1[v];
    repsOut[o + lm]       = y0;
    repsOut[o + 16 + lm]  = y1;
    repsHOut[o + lm]      = (_Float16)y0;
    repsHOut[o + 16 + lm] = (_Float16)y1;
  }
}

// ---------------------------------------------------------------------------
// Output: scalars (l=0 comp + invariant norms) -> masked pool -> W_top dot.
// ---------------------------------------------------------------------------
__global__ __launch_bounds__(256)
void kOut(const float* __restrict__ repsLv,   // [LVLS][B*N][16][C]
          const unsigned char* __restrict__ amask,
          const float* __restrict__ W_top, const float* __restrict__ b_top,
          float* __restrict__ out)
{
  int b = blockIdx.x, t = threadIdx.x;
  float acc = 0.0f;
  for (int idx = t; idx < N_ * 512; idx += 256) {
    int f = idx & 511, n = idx >> 9;
    int lvl = f >> 7, rem = f & 127, l = rem >> 5, c = rem & 31;
    const float* rep = repsLv + ((long)lvl * B_ * N_ + (long)(b * N_ + n)) * SLOTS_ * C_;
    float val;
    if (l == 0) val = rep[c];
    else {
      int sb = l * l, nm = 2 * l + 1;
      float s2 = 1e-12f;
      for (int mm = 0; mm < nm; ++mm) { float v = rep[(sb + mm) * C_ + c]; s2 += v * v; }
      val = sqrtf(s2);
    }
    acc += val * (float)amask[b * N_ + n] * W_top[f];
  }
  __shared__ float red[256];
  red[t] = acc; __syncthreads();
  for (int s = 128; s > 0; s >>= 1) { if (t < s) red[t] += red[t + s]; __syncthreads(); }
  if (t == 0) out[b] = red[0] + b_top[0];
}

// ---------------------------------------------------------------------------
extern "C" void kernel_launch(void* const* d_in, const int* in_sizes, int n_in,
                              void* d_out, int out_size, void* d_ws, size_t ws_size,
                              hipStream_t stream)
{
  const float*         positions = (const float*)d_in[0];
  const float*         one_hot   = (const float*)d_in[1];
  const int*           charges   = (const int*)d_in[2];
  const unsigned char* amask     = (const unsigned char*)d_in[3];  // jnp bool = 1 byte
  const unsigned char* emask     = (const unsigned char*)d_in[4];
  const float*         W_in      = (const float*)d_in[5];
  const float*         b_in      = (const float*)d_in[6];
  const float*         W_rad     = (const float*)d_in[7];
  const float*         W_edge    = (const float*)d_in[8];
  const float*         W_agg     = (const float*)d_in[9];
  const float*         W_self    = (const float*)d_in[10];
  const float*         W_top     = (const float*)d_in[11];
  const float*         b_top     = (const float*)d_in[12];
  float* out = (float*)d_out;

  // workspace carve-out (all 256B aligned)
  char* w = (char*)d_ws;
  size_t off = 0;
  auto carve = [&](size_t bytes) { void* p = w + off; off += (bytes + 255) & ~(size_t)255; return p; };
  _Float16* basisH = (_Float16*)carve((size_t)PAIRS_ * KPAD_ * 2);           // 8 MB
  float*    sphW   = (float*)   carve((size_t)PAIRS_ * SLOTS_ * 4);          // 4 MB
  float*    cut    = (float*)   carve((size_t)PAIRS_ * 4);                   // 256 KB
  float*    edge   = (float*)   carve((size_t)4 * PAIRS_ * C_ * 4);          // 32 MB
  size_t repStride = (size_t)B_ * N_ * SLOTS_ * C_;
  float*    reps   = (float*)   carve(repStride * (LVLS_ + 1) * 4);          // 10 MB (f32)
  _Float16* repsH  = (_Float16*)carve(repStride * (LVLS_ + 1) * 2);          // 5 MB (f16 shadow)
  _Float16* msgH   = (_Float16*)carve(repStride * 2);                        // 1 MB
  _Float16* wRadH  = (_Float16*)carve((size_t)16 * KPAD_ * C_ * 2);          // 64 KB (K-padded)
  _Float16* wEdgeH = (_Float16*)carve((size_t)16 * C_ * C_ * 2);             // 32 KB
  _Float16* wAggH  = (_Float16*)carve((size_t)16 * C_ * C_ * 2);             // 32 KB
  _Float16* wSelfH = (_Float16*)carve((size_t)16 * C_ * C_ * 2);             // 32 KB
  (void)ws_size; (void)in_sizes; (void)n_in; (void)out_size;

  kPrepAtoms<<<(B_ * N_ * C_ + 255) / 256, 256, 0, stream>>>(
      one_hot, charges, amask, W_in, b_in, reps, repsH);
  kPrepW<<<(16 * KPAD_ * C_ + 255) / 256, 256, 0, stream>>>(
      W_rad, W_edge, W_agg, W_self, wRadH, wEdgeH, wAggH, wSelfH);
  kPrepPairs<<<(int)((PAIRS_ + 255) / 256), 256, 0, stream>>>(
      positions, emask, basisH, sphW, cut, edge);

  for (int lvl = 0; lvl < LVLS_; ++lvl) {
    const _Float16* repsHIn = repsH + (size_t)lvl * repStride;
    float*          repsOut = reps  + (size_t)(lvl + 1) * repStride;
    _Float16*       repsHOut= repsH + (size_t)(lvl + 1) * repStride;
    kEdge<<<B_ * N_ * 4, 32, 0, stream>>>(repsHIn, basisH, cut, edge, wRadH, wEdgeH, lvl);
    kMsg <<<B_ * N_, 128, 0, stream>>>(edge, sphW, amask, msgH);
    kRep <<<1024, 32, 0, stream>>>(msgH, repsHIn, repsOut, repsHOut, wAggH, wSelfH, amask, lvl);
  }
  kOut<<<B_, 256, 0, stream>>>(reps + repStride, amask, W_top, b_top, out);
}
